// multihead_attention_77506979823706
// MI455X (gfx1250) — compile-verified, matrix operand order checked by reading, not run on hardware
//
#include <hip/hip_runtime.h>
#include <hip/hip_bf16.h>
#include <math.h>

// ---------------------------------------------------------------------------
// MHA forward on gfx1250: f16 WMMA + async-to-LDS staging (ASYNCcnt path).
// B=2, S=2048, D=1024, H=16, hd=64.
// ---------------------------------------------------------------------------

typedef _Float16 half16 __attribute__((ext_vector_type(16)));
typedef float    float8 __attribute__((ext_vector_type(8)));

#define D_MODEL   1024
#define SEQ       2048
#define NHEADS    16
#define HDIM      64
#define BATCH     2
#define MTOT      (BATCH * SEQ)          // 4096 rows of x

__device__ __forceinline__ float8 wmma_f16f32(half16 a, half16 b, float8 c) {
  return __builtin_amdgcn_wmma_f32_16x16x32_f16(
      false, a, false, b, (short)0, c, false, false);
}

// Low 32 bits of a generic pointer to LDS == LDS byte offset (ISA 10.2).
__device__ __forceinline__ unsigned lds_off(const void* p) {
  return (unsigned)(unsigned long long)p;
}

// Async global -> LDS, 16B per lane, GV mode. Tracked by ASYNCcnt.
__device__ __forceinline__ void async_b128(void* lds_dst, const void* g) {
  asm volatile("global_load_async_to_lds_b128 %0, %1, off"
               :: "v"(lds_off(lds_dst)), "v"(g)
               : "memory");
}
__device__ __forceinline__ void wait_async0() { asm volatile("s_wait_asynccnt 0x0" ::: "memory"); }
__device__ __forceinline__ void wait_async1() { asm volatile("s_wait_asynccnt 0x1" ::: "memory"); }
__device__ __forceinline__ void wait_async2() { asm volatile("s_wait_asynccnt 0x2" ::: "memory"); }

// ---------------------------------------------------------------------------
// f32 -> f16 conversion (4 elements / thread; n is a multiple of 1024)
// ---------------------------------------------------------------------------
__global__ __launch_bounds__(256)
void cvt_f32_f16(const float* __restrict__ s, _Float16* __restrict__ d, int n) {
  int i = (blockIdx.x * blockDim.x + threadIdx.x) * 4;
  if (i + 3 < n) {
    float4 v = *(const float4*)(s + i);
    d[i + 0] = (_Float16)v.x;
    d[i + 1] = (_Float16)v.y;
    d[i + 2] = (_Float16)v.z;
    d[i + 3] = (_Float16)v.w;
  }
}

// ---------------------------------------------------------------------------
// Block-cooperative GEMM core: block = 8 waves = 128(M) x 64(N) tile.
// W tile (64 N-rows x 32 K-halves = 4KB) staged in LDS via async loads,
// double buffered; each wave computes a 16x64 sub-tile (4 WMMA accs).
// A rows are per-wave from global (L2-resident).
// store(gm, gn, v) writes one output element.
// ---------------------------------------------------------------------------
template <typename StoreFn>
__device__ __forceinline__ void proj_core(const _Float16* __restrict__ xh,
                                          const _Float16* __restrict__ wh,
                                          const float* __restrict__ bias,
                                          StoreFn store) {
  __shared__ __align__(128) _Float16 wsh[2][64][32];
  const int tid   = threadIdx.x;
  const int lane  = tid & 31;
  const int lrow  = lane & 15;
  const int lhalf = lane >> 4;
  const int wib   = tid >> 5;
  const int mbase = (blockIdx.x >> 4) * 128 + wib * 16;
  const int nbase = (blockIdx.x & 15) * 64;

  // staging map: 64 rows x 64B; 4 lanes of 16B per row
  const int srow = tid >> 2;
  const int schk = (tid & 3) * 8;  // halves
  const _Float16* gsrc = wh + (size_t)(nbase + srow) * D_MODEL + schk;

  async_b128(&wsh[0][srow][schk], gsrc + 0);
  async_b128(&wsh[1][srow][schk], gsrc + 32);

  float8 acc[4] = {};
  const _Float16* arow = xh + (size_t)(mbase + lrow) * D_MODEL + lhalf * 16;

  const int NSTEP = D_MODEL / 32;
  for (int i = 0; i < NSTEP; ++i) {
    if (i + 1 < NSTEP) wait_async1(); else wait_async0();
    __syncthreads();
    const int buf = i & 1;
    half16 a = *(const half16*)(arow + i * 32);
    __builtin_prefetch(arow + i * 32 + 64, 0, 0);
#pragma unroll
    for (int t = 0; t < 4; ++t) {
      half16 b = *(const half16*)(&wsh[buf][t * 16 + lrow][lhalf * 16]);
      acc[t] = wmma_f16f32(a, b, acc[t]);
    }
    __syncthreads();
    if (i + 2 < NSTEP)
      async_b128(&wsh[buf][srow][schk], gsrc + (i + 2) * 32);
  }

#pragma unroll
  for (int t = 0; t < 4; ++t) {
    const int gn = nbase + t * 16 + lrow;
    const float bv = bias[gn];
#pragma unroll
    for (int r = 0; r < 8; ++r) {
      const int gm = mbase + r + 8 * lhalf;
      store(gm, gn, acc[t][r] + bv);
    }
  }
}

// QKV projection: vmode=0 -> [B,H,S,hd] (Q,K); vmode=1 -> [B,H,hd,S] (V^T)
__global__ __launch_bounds__(256)
void proj_kernel(const _Float16* __restrict__ xh, const _Float16* __restrict__ wh,
                 const float* __restrict__ bias, _Float16* __restrict__ out,
                 int vmode) {
  proj_core(xh, wh, bias, [=](int gm, int gn, float v) {
    const int b = gm >> 11;
    const int s = gm & 2047;
    const int h = gn >> 6;
    const int d = gn & 63;
    if (vmode)
      out[(((size_t)(b * NHEADS + h) * HDIM + d) * SEQ) + s] = (_Float16)v;
    else
      out[(((size_t)(b * NHEADS + h) * SEQ + s) * HDIM) + d] = (_Float16)v;
  });
}

// Output projection: f32 stores straight to d_out [B*S, D].
__global__ __launch_bounds__(256)
void oproj_kernel(const _Float16* __restrict__ xh, const _Float16* __restrict__ wh,
                  const float* __restrict__ bias, float* __restrict__ out) {
  proj_core(xh, wh, bias, [=](int gm, int gn, float v) {
    out[(size_t)gm * D_MODEL + gn] = v;
  });
}

// ---------------------------------------------------------------------------
// Flash attention, block-cooperative: 8 waves of a block handle 8 q-tiles of
// the SAME (b,h), sharing K (32x64) and V^T (64x32) tiles staged in LDS via
// async loads (double buffered). One wave = one 16-query tile.
// q,k: [B,H,S,hd] f16; vt: [B,H,hd,S] f16; out: [B,S,D] f16.
// grid: 512 blocks (32 bh * 16 q-groups) x 256 threads.
// ---------------------------------------------------------------------------
__global__ __launch_bounds__(256)
void attn_kernel(const _Float16* __restrict__ qb, const _Float16* __restrict__ kb,
                 const _Float16* __restrict__ vtb, _Float16* __restrict__ ob) {
  __shared__ __align__(128) _Float16 ksh[2][32][64];   // [buf][key][dim]
  __shared__ __align__(128) _Float16 vsh[2][64][32];   // [buf][dim][key]
  __shared__ __align__(64)  _Float16 psh[8][16][32];   // per-wave P repack

  const int tid   = threadIdx.x;
  const int lane  = tid & 31;
  const int lrow  = lane & 15;
  const int lhalf = lane >> 4;
  const int wib   = tid >> 5;
  const int bh    = blockIdx.x >> 4;              // 0..31
  const int qt    = (blockIdx.x & 15) * 8 + wib;  // 0..127

  const _Float16* qbase = qb  + (size_t)bh * SEQ * HDIM;
  const _Float16* kbase = kb  + (size_t)bh * SEQ * HDIM;
  const _Float16* vbase = vtb + (size_t)bh * HDIM * SEQ;

  // staging maps (256 lanes x 16B = 4KB per tile per step)
  const int krow = tid >> 3, kchk = (tid & 7) * 8;  // ksh rows: 128B
  const int vrow = tid >> 2, vchk = (tid & 3) * 8;  // vsh rows: 64B
  const _Float16* kg = kbase + (size_t)krow * HDIM + kchk;
  const _Float16* vg = vbase + (size_t)vrow * SEQ + vchk;

  async_b128(&ksh[0][krow][kchk], kg);
  async_b128(&vsh[0][vrow][vchk], vg);
  async_b128(&ksh[1][krow][kchk], kg + 32 * HDIM);
  async_b128(&vsh[1][vrow][vchk], vg + 32);

  // Q fragments (A layout): row = qt*16 + l%16, K-chunks [0..31], [32..63]
  half16 qa0 = *(const half16*)(qbase + (size_t)(qt * 16 + lrow) * HDIM + lhalf * 16);
  half16 qa1 = *(const half16*)(qbase + (size_t)(qt * 16 + lrow) * HDIM + 32 + lhalf * 16);

  float m[8], l[8];
  float8 o[4] = {};
#pragma unroll
  for (int r = 0; r < 8; ++r) { m[r] = -1e30f; l[r] = 0.0f; }

  const float sc = 0.125f;                 // 1/sqrt(64)
  const int NSTEP = SEQ / 32;              // 64 key-steps

  for (int i = 0; i < NSTEP; ++i) {
    if (i + 1 < NSTEP) wait_async2(); else wait_async0();
    __syncthreads();
    const int buf = i & 1;

    // ---- scores: two 16x16 tiles, each = 2 chained K=32 WMMAs over hd=64
    float8 z = {};
    half16 kf00 = *(const half16*)(&ksh[buf][lrow][lhalf * 16]);
    half16 kf01 = *(const half16*)(&ksh[buf][lrow][32 + lhalf * 16]);
    float8 s0 = wmma_f16f32(qa0, kf00, z);
    s0 = wmma_f16f32(qa1, kf01, s0);
    half16 kf10 = *(const half16*)(&ksh[buf][16 + lrow][lhalf * 16]);
    half16 kf11 = *(const half16*)(&ksh[buf][16 + lrow][32 + lhalf * 16]);
    float8 s1 = wmma_f16f32(qa0, kf10, z);
    s1 = wmma_f16f32(qa1, kf11, s1);

    // ---- online softmax (rows r+8*lhalf live in this lane's half-wave)
    float mt[8];
#pragma unroll
    for (int r = 0; r < 8; ++r) mt[r] = fmaxf(s0[r], s1[r]) * sc;
#pragma unroll
    for (int mask = 1; mask <= 8; mask <<= 1) {
#pragma unroll
      for (int r = 0; r < 8; ++r)
        mt[r] = fmaxf(mt[r], __shfl_xor(mt[r], mask, 32));
    }
    float alpha[8], p0[8], p1[8], rs[8];
#pragma unroll
    for (int r = 0; r < 8; ++r) {
      const float mn = fmaxf(m[r], mt[r]);
      alpha[r] = __expf(m[r] - mn);
      p0[r]    = __expf(s0[r] * sc - mn);
      p1[r]    = __expf(s1[r] * sc - mn);
      m[r]     = mn;
      rs[r]    = p0[r] + p1[r];
    }
#pragma unroll
    for (int mask = 1; mask <= 8; mask <<= 1) {
#pragma unroll
      for (int r = 0; r < 8; ++r)
        rs[r] += __shfl_xor(rs[r], mask, 32);
    }
#pragma unroll
    for (int r = 0; r < 8; ++r) l[r] = l[r] * alpha[r] + rs[r];
#pragma unroll
    for (int t = 0; t < 4; ++t)
#pragma unroll
      for (int r = 0; r < 8; ++r) o[t][r] *= alpha[r];

    // ---- repack P (C layout -> A layout) through per-wave LDS tile
#pragma unroll
    for (int r = 0; r < 8; ++r) {
      psh[wib][r + 8 * lhalf][lrow]      = (_Float16)p0[r];
      psh[wib][r + 8 * lhalf][16 + lrow] = (_Float16)p1[r];
    }
    __builtin_amdgcn_wave_barrier();
    half16 pa = *(const half16*)(&psh[wib][lrow][lhalf * 16]);
    __builtin_amdgcn_wave_barrier();

    // ---- O += P * V  (B frag from LDS V^T: contiguous key runs)
#pragma unroll
    for (int t = 0; t < 4; ++t) {
      half16 vf = *(const half16*)(&vsh[buf][t * 16 + lrow][lhalf * 16]);
      o[t] = wmma_f16f32(pa, vf, o[t]);
    }

    __syncthreads();   // all reads of buf done before overwrite
    if (i + 2 < NSTEP) {
      async_b128(&ksh[buf][krow][kchk], kg + (size_t)(i + 2) * 32 * HDIM);
      async_b128(&vsh[buf][vrow][vchk], vg + (i + 2) * 32);
    }
  }

  // ---- normalize and store O as f16 [B,S,D]
  const int b = bh >> 4, h = bh & 15;
#pragma unroll
  for (int t = 0; t < 4; ++t) {
#pragma unroll
    for (int r = 0; r < 8; ++r) {
      const int s = qt * 16 + r + 8 * lhalf;
      const float v = o[t][r] / l[r];
      ob[((size_t)(b * SEQ + s)) * D_MODEL + h * HDIM + t * 16 + lrow] = (_Float16)v;
    }
  }
}

// ---------------------------------------------------------------------------
// Host-side launcher
// ---------------------------------------------------------------------------
extern "C" void kernel_launch(void* const* d_in, const int* in_sizes, int n_in,
                              void* d_out, int out_size, void* d_ws, size_t ws_size,
                              hipStream_t stream) {
  (void)in_sizes; (void)n_in; (void)out_size; (void)ws_size;

  const float* x  = (const float*)d_in[0];
  const float* Wq = (const float*)d_in[1];
  const float* bq = (const float*)d_in[2];
  const float* Wk = (const float*)d_in[3];
  const float* bk = (const float*)d_in[4];
  const float* Wv = (const float*)d_in[5];
  const float* bv = (const float*)d_in[6];
  const float* Wo = (const float*)d_in[7];
  const float* bo = (const float*)d_in[8];
  float* out = (float*)d_out;

  char* ws = (char*)d_ws;
  const size_t MB = 1024ull * 1024ull;
  _Float16* xh  = (_Float16*)(ws + 0 * MB);    // 8 MB  [4096,1024]
  _Float16* wqh = (_Float16*)(ws + 8 * MB);    // 2 MB
  _Float16* wkh = (_Float16*)(ws + 10 * MB);   // 2 MB
  _Float16* wvh = (_Float16*)(ws + 12 * MB);   // 2 MB
  _Float16* woh = (_Float16*)(ws + 14 * MB);   // 2 MB
  _Float16* qb  = (_Float16*)(ws + 16 * MB);   // 8 MB  [B,H,S,hd]
  _Float16* kb  = (_Float16*)(ws + 24 * MB);   // 8 MB  [B,H,S,hd]
  _Float16* vtb = (_Float16*)(ws + 32 * MB);   // 8 MB  [B,H,hd,S]
  _Float16* ob  = (_Float16*)(ws + 40 * MB);   // 8 MB  [B,S,D]

  const int nx = MTOT * D_MODEL;
  const int nw = D_MODEL * D_MODEL;

  cvt_f32_f16<<<nx / 1024, 256, 0, stream>>>(x,  xh,  nx);
  cvt_f32_f16<<<nw / 1024, 256, 0, stream>>>(Wq, wqh, nw);
  cvt_f32_f16<<<nw / 1024, 256, 0, stream>>>(Wk, wkh, nw);
  cvt_f32_f16<<<nw / 1024, 256, 0, stream>>>(Wv, wvh, nw);
  cvt_f32_f16<<<nw / 1024, 256, 0, stream>>>(Wo, woh, nw);

  proj_kernel<<<512, 256, 0, stream>>>(xh, wqh, bq, qb,  0);
  proj_kernel<<<512, 256, 0, stream>>>(xh, wkh, bk, kb,  0);
  proj_kernel<<<512, 256, 0, stream>>>(xh, wvh, bv, vtb, 1);

  attn_kernel<<<512, 256, 0, stream>>>(qb, kb, vtb, ob);

  oproj_kernel<<<512, 256, 0, stream>>>(ob, woh, bo, out);
}
